// LEConvNet_88553635709227
// MI455X (gfx1250) — matrix operand order repaired and checked by reading, not compiled, not run on hardware
//
#include <hip/hip_runtime.h>
#include <math.h>

// ---------------------------------------------------------------------------
// Types for CDNA5 WMMA (gfx1250, wave32)
// ---------------------------------------------------------------------------
typedef unsigned short u16;
typedef __attribute__((ext_vector_type(8)))  u16    u16x8;
typedef __attribute__((ext_vector_type(16))) u16    u16x16;
typedef __attribute__((ext_vector_type(16))) __bf16 v16bf;
typedef __attribute__((ext_vector_type(8)))  float  v8f;

__device__ __forceinline__ u16 f2bf(float f) {
    unsigned u = __builtin_bit_cast(unsigned, f);
    unsigned r = u + 0x7FFFu + ((u >> 16) & 1u);   // round-to-nearest-even
    return (u16)(r >> 16);
}

// ---------------------------------------------------------------------------
// f32 -> bf16 bulk convert (vectorized by 4)
// ---------------------------------------------------------------------------
__global__ void f32_to_bf16_k(const float* __restrict__ in, u16* __restrict__ out,
                              long long n4) {
    long long i = ((long long)blockIdx.x * blockDim.x + threadIdx.x);
    if (i >= n4) return;
    const float4 v = *(const float4*)(in + i * 4);
    u16* o = out + i * 4;
    o[0] = f2bf(v.x); o[1] = f2bf(v.y); o[2] = f2bf(v.z); o[3] = f2bf(v.w);
}

// ---------------------------------------------------------------------------
// Pack weights W[K x Cout] (f32, row-major) into the per-lane B-fragment
// layout of V_WMMA_F32_16X16X32_BF16:
//   element i of lane L (lo=L&15, hi=L>>4) holds  B[k = kt*32 + 16*(i>>3) + 8*hi + (i&7)][n = nt*16 + lo]
// Packed index: p = ((kt*nT + nt)*32 + lane)*16 + i   -> one 32B load per lane.
// ---------------------------------------------------------------------------
__global__ void pack_w_k(const float* __restrict__ W, u16* __restrict__ Wp,
                         int K, int Cout, int nT, int total) {
    int p = blockIdx.x * blockDim.x + threadIdx.x;
    if (p >= total) return;
    int i    = p & 15;
    int lane = (p >> 4) & 31;
    int tile = p >> 9;
    int nt = tile % nT, kt = tile / nT;
    int hi = lane >> 4, lo = lane & 15;
    int k = kt * 32 + (i >> 3) * 16 + hi * 8 + (i & 7);
    int n = nt * 16 + lo;
    float v = (k < K && n < Cout) ? W[(size_t)k * Cout + n] : 0.0f;
    Wp[p] = f2bf(v);
}

// ---------------------------------------------------------------------------
// WMMA GEMM:  out[Nrows x Cout] = A[Nrows x K](bf16) @ Wp(packed bf16) + bias
// Block = 256 threads = 8 waves; each wave owns one 16-row tile.
// MODE 0: store f32 (+bias).  MODE 1: store bf16(elu(f32 + bias)).
// Row guard is wave-uniform -> EXEC stays all-ones around v_wmma (ISA req).
// ---------------------------------------------------------------------------
template <int NT, int MODE>
__global__ __launch_bounds__(256)
void gemm_wmma_k(const u16* __restrict__ A, const u16* __restrict__ Wp,
                 const float* __restrict__ bias,
                 float* __restrict__ outF, u16* __restrict__ outBF,
                 int Nrows, int K, int Cout) {
    const int wave = threadIdx.x >> 5;
    const int lane = threadIdx.x & 31;
    const int lo = lane & 15, hi = lane >> 4;
    const int rowTile = blockIdx.x * 8 + wave;
    if (rowTile * 16 >= Nrows) return;            // wave-uniform exit
    const int rowBase = rowTile << 4;
    const int kTiles  = K >> 5;                   // K is a multiple of 32 here

    v8f acc[NT] = {};

    for (int kt = 0; kt < kTiles; ++kt) {
        // A fragment: element i -> k = kt*32 + 16*(i>>3) + 8*hi + (i&7)
        const u16* ap = A + (size_t)(rowBase + lo) * K + kt * 32 + hi * 8;
        u16x8 alo = *(const u16x8*)(ap);
        u16x8 ahi = *(const u16x8*)(ap + 16);
        u16x16 au = __builtin_shufflevector(alo, ahi,
                                            0, 1, 2, 3, 4, 5, 6, 7,
                                            8, 9, 10, 11, 12, 13, 14, 15);
        v16bf afrag = __builtin_bit_cast(v16bf, au);

#pragma unroll
        for (int nt = 0; nt < NT; ++nt) {
            const u16* bp = Wp + (((size_t)(kt * NT + nt) * 32 + lane) << 4);
            v16bf bfrag = __builtin_bit_cast(v16bf, *(const u16x16*)bp);
            acc[nt] = __builtin_amdgcn_wmma_f32_16x16x32_bf16(
                false, afrag, false, bfrag, (short)0, acc[nt], false, false);
        }
    }

    // Epilogue: C/D layout -> element r of lane = (m = rowBase + r + 8*hi, n = lo)
#pragma unroll
    for (int nt = 0; nt < NT; ++nt) {
        int n = nt * 16 + lo;
        if (n < Cout) {
            float bv = bias ? bias[n] : 0.0f;
#pragma unroll
            for (int r = 0; r < 8; ++r) {
                int m = rowBase + r + 8 * hi;
                float v = acc[nt][r] + bv;
                if (MODE == 0) {
                    outF[(size_t)m * Cout + n] = v;
                } else {
                    float e = v > 0.0f ? v : expm1f(v);
                    outBF[(size_t)m * Cout + n] = f2bf(e);
                }
            }
        }
    }
}

// ---------------------------------------------------------------------------
// Edge scatter for LEConv:  agg[dst] += ew * (a[src] - b[dst])
// One thread per (edge, 4-channel quad). agg was pre-initialized to lin3 out.
// All gather tables are L2-resident (<= 25.6 MB vs 192 MB L2).
// ---------------------------------------------------------------------------
__global__ __launch_bounds__(256)
void scatter_leconv_k(const int* __restrict__ src, const int* __restrict__ dst,
                      const float* __restrict__ ew,
                      const float* __restrict__ a, const float* __restrict__ b,
                      float* __restrict__ agg,
                      long long E, int qshift, int C) {
    long long gid = (long long)blockIdx.x * blockDim.x + threadIdx.x;
    if (gid >= (E << qshift)) return;
    long long e = gid >> qshift;
    int q = (int)(gid & ((1 << qshift) - 1));
    int s = src[e], d = dst[e];
    float w = ew[e];
    const float4 av = *(const float4*)(a + (size_t)s * C + q * 4);
    const float4 bv = *(const float4*)(b + (size_t)d * C + q * 4);
    float* ag = agg + (size_t)d * C + q * 4;
    atomicAdd(ag + 0, w * (av.x - bv.x));
    atomicAdd(ag + 1, w * (av.y - bv.y));
    atomicAdd(ag + 2, w * (av.z - bv.z));
    atomicAdd(ag + 3, w * (av.w - bv.w));
}

// ---------------------------------------------------------------------------
// ELU (f32) -> bf16
// ---------------------------------------------------------------------------
__global__ void elu_to_bf16_k(const float* __restrict__ in, u16* __restrict__ out,
                              long long n) {
    long long i = (long long)blockIdx.x * blockDim.x + threadIdx.x;
    if (i >= n) return;
    float v = in[i];
    out[i] = f2bf(v > 0.0f ? v : expm1f(v));
}

// ---------------------------------------------------------------------------
// Row-wise log_softmax (NC=10)
// ---------------------------------------------------------------------------
__global__ void logsoftmax_k(const float* __restrict__ logits, float* __restrict__ out,
                             int N, int NC) {
    int i = blockIdx.x * blockDim.x + threadIdx.x;
    if (i >= N) return;
    const float* r = logits + (size_t)i * NC;
    float mx = r[0];
    for (int c = 1; c < NC; ++c) mx = fmaxf(mx, r[c]);
    float s = 0.0f;
    for (int c = 0; c < NC; ++c) s += expf(r[c] - mx);
    float l = logf(s);
    float* o = out + (size_t)i * NC;
    for (int c = 0; c < NC; ++c) o[c] = r[c] - mx - l;
}

// ---------------------------------------------------------------------------
// Host-side orchestration
// ---------------------------------------------------------------------------
static inline unsigned cdiv(long long a, long long b) { return (unsigned)((a + b - 1) / b); }

static void gemm_dispatch(int NT, int MODE,
                          const u16* A, const u16* Wp, const float* bias,
                          float* outF, u16* outBF, int Nrows, int K, int Cout,
                          hipStream_t stream) {
    int rowTiles = (Nrows + 15) / 16;
    unsigned gb = (unsigned)((rowTiles + 7) / 8);
    if (NT == 1 && MODE == 0)
        gemm_wmma_k<1, 0><<<gb, 256, 0, stream>>>(A, Wp, bias, outF, outBF, Nrows, K, Cout);
    else if (NT == 2 && MODE == 0)
        gemm_wmma_k<2, 0><<<gb, 256, 0, stream>>>(A, Wp, bias, outF, outBF, Nrows, K, Cout);
    else if (NT == 4 && MODE == 0)
        gemm_wmma_k<4, 0><<<gb, 256, 0, stream>>>(A, Wp, bias, outF, outBF, Nrows, K, Cout);
    else if (NT == 8 && MODE == 1)
        gemm_wmma_k<8, 1><<<gb, 256, 0, stream>>>(A, Wp, bias, outF, outBF, Nrows, K, Cout);
}

static void pack_w(const float* W, u16* Wp, int K, int Cout, hipStream_t stream) {
    int nT = (Cout + 15) >> 4;
    int kT = K >> 5;
    int total = kT * nT * 512;
    pack_w_k<<<cdiv(total, 256), 256, 0, stream>>>(W, Wp, K, Cout, nT, total);
}

extern "C" void kernel_launch(void* const* d_in, const int* in_sizes, int n_in,
                              void* d_out, int out_size, void* d_ws, size_t ws_size,
                              hipStream_t stream) {
    const int F_IN = 256, C1 = 32, C2 = 64, H = 128, NC = 10;
    const int N = in_sizes[0] / F_IN;
    const long long E = in_sizes[1] / 2;

    const float* x    = (const float*)d_in[0];
    const int*   ei   = (const int*)d_in[1];
    const float* ea   = (const float*)d_in[2];
    const float* W1a  = (const float*)d_in[3];
    const float* b1a  = (const float*)d_in[4];
    const float* W1b  = (const float*)d_in[5];
    const float* W1c  = (const float*)d_in[6];
    const float* b1c  = (const float*)d_in[7];
    const float* W2a  = (const float*)d_in[8];
    const float* b2a  = (const float*)d_in[9];
    const float* W2b  = (const float*)d_in[10];
    const float* W2c  = (const float*)d_in[11];
    const float* b2c  = (const float*)d_in[12];
    const float* Wf1  = (const float*)d_in[13];
    const float* bf1  = (const float*)d_in[14];
    const float* Wf2  = (const float*)d_in[15];
    const float* bf2  = (const float*)d_in[16];

    const int* src = ei;
    const int* dst = ei + E;

    // ---- carve workspace (all offsets 256B aligned) ----
    char* base = (char*)d_ws;
    size_t off = 0;
    auto alloc = [&](size_t bytes) -> void* {
        void* p = base + off;
        off = (off + bytes + 255) & ~(size_t)255;
        return p;
    };
    u16*   xh     = (u16*)  alloc((size_t)N * F_IN * sizeof(u16));
    float* a1     = (float*)alloc((size_t)N * C1 * sizeof(float));
    float* b1     = (float*)alloc((size_t)N * C1 * sizeof(float));
    float* agg1   = (float*)alloc((size_t)N * C1 * sizeof(float));
    u16*   h1bf   = (u16*)  alloc((size_t)N * C1 * sizeof(u16));
    float* a2     = (float*)alloc((size_t)N * C2 * sizeof(float));
    float* b2     = (float*)alloc((size_t)N * C2 * sizeof(float));
    float* agg2   = (float*)alloc((size_t)N * C2 * sizeof(float));
    u16*   h2bf   = (u16*)  alloc((size_t)N * C2 * sizeof(u16));
    u16*   h3bf   = (u16*)  alloc((size_t)N * H  * sizeof(u16));
    float* logits = (float*)alloc((size_t)N * NC * sizeof(float));
    u16*   Wp1a   = (u16*)  alloc((size_t)F_IN * 32  * sizeof(u16));
    u16*   Wp1b   = (u16*)  alloc((size_t)F_IN * 32  * sizeof(u16));
    u16*   Wp1c   = (u16*)  alloc((size_t)F_IN * 32  * sizeof(u16));
    u16*   Wp2a   = (u16*)  alloc((size_t)C1 * 64    * sizeof(u16));
    u16*   Wp2b   = (u16*)  alloc((size_t)C1 * 64    * sizeof(u16));
    u16*   Wp2c   = (u16*)  alloc((size_t)C1 * 64    * sizeof(u16));
    u16*   Wpf1   = (u16*)  alloc((size_t)C2 * 128   * sizeof(u16));
    u16*   Wpf2   = (u16*)  alloc((size_t)H  * 16    * sizeof(u16));
    (void)ws_size;

    // ---- prep: bf16 copy of x, packed bf16 weights ----
    {
        long long n4 = (long long)N * F_IN / 4;
        f32_to_bf16_k<<<cdiv(n4, 256), 256, 0, stream>>>(x, xh, n4);
    }
    pack_w(W1a, Wp1a, F_IN, C1, stream);
    pack_w(W1b, Wp1b, F_IN, C1, stream);
    pack_w(W1c, Wp1c, F_IN, C1, stream);
    pack_w(W2a, Wp2a, C1, C2, stream);
    pack_w(W2b, Wp2b, C1, C2, stream);
    pack_w(W2c, Wp2c, C1, C2, stream);
    pack_w(Wf1, Wpf1, C2, H, stream);
    pack_w(Wf2, Wpf2, H, NC, stream);

    // ---- conv1: a1 = x@W1a+b1a, b1 = x@W1b, agg1 = x@W1c+b1c ----
    gemm_dispatch(2, 0, xh, Wp1a, b1a, a1,   nullptr, N, F_IN, C1, stream);
    gemm_dispatch(2, 0, xh, Wp1b, nullptr, b1, nullptr, N, F_IN, C1, stream);
    gemm_dispatch(2, 0, xh, Wp1c, b1c, agg1, nullptr, N, F_IN, C1, stream);
    // scatter: agg1 += ew * (a1[src] - b1[dst])
    {
        long long total = E * (C1 / 4);
        scatter_leconv_k<<<cdiv(total, 256), 256, 0, stream>>>(
            src, dst, ea, a1, b1, agg1, E, 3 /*log2(32/4)*/, C1);
    }
    elu_to_bf16_k<<<cdiv((long long)N * C1, 256), 256, 0, stream>>>(agg1, h1bf, (long long)N * C1);

    // ---- conv2 ----
    gemm_dispatch(4, 0, h1bf, Wp2a, b2a, a2,   nullptr, N, C1, C2, stream);
    gemm_dispatch(4, 0, h1bf, Wp2b, nullptr, b2, nullptr, N, C1, C2, stream);
    gemm_dispatch(4, 0, h1bf, Wp2c, b2c, agg2, nullptr, N, C1, C2, stream);
    {
        long long total = E * (C2 / 4);
        scatter_leconv_k<<<cdiv(total, 256), 256, 0, stream>>>(
            src, dst, ea, a2, b2, agg2, E, 4 /*log2(64/4)*/, C2);
    }
    elu_to_bf16_k<<<cdiv((long long)N * C2, 256), 256, 0, stream>>>(agg2, h2bf, (long long)N * C2);

    // ---- fc1 (fused elu -> bf16), fc2 -> logits ----
    gemm_dispatch(8, 1, h2bf, Wpf1, bf1, nullptr, h3bf, N, C2, H, stream);
    gemm_dispatch(1, 0, h3bf, Wpf2, bf2, logits, nullptr, N, H, NC, stream);

    // ---- log_softmax -> d_out ----
    logsoftmax_k<<<cdiv(N, 256), 256, 0, stream>>>(logits, (float*)d_out, N, NC);
}
